// SubGcn_16080357556240
// MI455X (gfx1250) — matrix operand order repaired and checked by reading, not compiled, not run on hardware
//
#include <hip/hip_runtime.h>
#include <hip/hip_bf16.h>

typedef __attribute__((ext_vector_type(16))) _Float16 v16h;
typedef __attribute__((ext_vector_type(8)))  _Float16 v8h;
typedef __attribute__((ext_vector_type(8)))  float    v8f;

#define N_GRAPHS 256
#define HID 64

// ---------------- utility ----------------
__global__ void zero_f32(float* __restrict__ p, int n) {
  int i = blockIdx.x * blockDim.x + threadIdx.x;
  if (i < n) p[i] = 0.0f;
}

// deg accumulation: norm[dst[e]] += 1
__global__ void deg_kernel(const int* __restrict__ dst, float* __restrict__ nrm, int E) {
  int e = blockIdx.x * blockDim.x + threadIdx.x;
  if (e < E) unsafeAtomicAdd(&nrm[dst[e]], 1.0f);
}

// norm = (deg + 1)^-0.5  (in place)
__global__ void norm_kernel(float* __restrict__ nrm, int N) {
  int i = blockIdx.x * blockDim.x + threadIdx.x;
  if (i < N) nrm[i] = rsqrtf(nrm[i] + 1.0f);
}

// scatter: agg[dst] += norm[src] * h[src]; 4 threads per edge, 8 channels each
__global__ void scatter_kernel(const int* __restrict__ src, const int* __restrict__ dst,
                               const float* __restrict__ nrm,
                               const float* __restrict__ hin,
                               float* __restrict__ agg, int E) {
  int t = blockIdx.x * blockDim.x + threadIdx.x;
  int e = t >> 2;
  if (e >= E) return;
  int cg = (t & 3) * 8;
  int s = src[e];
  int d = dst[e];
  float ns = nrm[s];
  const float4* hp = (const float4*)(hin + (size_t)s * 32 + cg);
  float4 a = hp[0];
  float4 b = hp[1];
  float* ap = agg + (size_t)d * 32 + cg;
  unsafeAtomicAdd(ap + 0, a.x * ns);
  unsafeAtomicAdd(ap + 1, a.y * ns);
  unsafeAtomicAdd(ap + 2, a.z * ns);
  unsafeAtomicAdd(ap + 3, a.w * ns);
  unsafeAtomicAdd(ap + 4, b.x * ns);
  unsafeAtomicAdd(ap + 5, b.y * ns);
  unsafeAtomicAdd(ap + 6, b.z * ns);
  unsafeAtomicAdd(ap + 7, b.w * ns);
}

// h_out = norm*agg + norm^2*h_in   (elementwise over N*32)
__global__ void combine_kernel(const float* __restrict__ nrm,
                               const float* __restrict__ agg,
                               const float* __restrict__ hin,
                               float* __restrict__ hout, int total) {
  int i = blockIdx.x * blockDim.x + threadIdx.x;
  if (i >= total) return;
  float nv = nrm[i >> 5];
  hout[i] = nv * agg[i] + nv * nv * hin[i];
}

// f32 -> f16 conversion
__global__ void cvt_f16_kernel(const float* __restrict__ in, _Float16* __restrict__ out, int n) {
  int i = blockIdx.x * blockDim.x + threadIdx.x;
  if (i < n) out[i] = (_Float16)in[i];
}

// per-graph node counts
__global__ void counts_kernel(const int* __restrict__ batch, float* __restrict__ counts, int N) {
  int i = blockIdx.x * blockDim.x + threadIdx.x;
  if (i < N) unsafeAtomicAdd(&counts[batch[i]], 1.0f);
}

// WMMA GEMM: relu(h[N,32] @ w.T[32,64] + b) fused with mean-pool sum accumulation.
// One wave per (16-row, 16-outch) tile; blockDim=128 -> 4 waves cover all 64 out chans.
__global__ void gcn_gemm_relu_pool(const _Float16* __restrict__ hf,   // [N,32]
                                   const _Float16* __restrict__ wf,   // [64,32] (outch-major)
                                   const float*    __restrict__ bias, // [64]
                                   const int*      __restrict__ batch,// [N]
                                   float*          __restrict__ pool, // [256,64]
                                   int N) {
  int lane = threadIdx.x & 31;
  int wave = threadIdx.x >> 5;          // n-tile: 0..3
  int row0 = blockIdx.x * 16;
  int m  = lane & 15;
  int hi = lane >> 4;

  // ---- A tile: 16x32 f16.  lane holds row m, K = kb..kb+7 and kb+16..kb+23
  int arow = row0 + m;
  if (arow >= N) arow = N - 1;          // N is a multiple of 16; defensive clamp only
  int kb = hi * 8;
  const v8h* ap0 = (const v8h*)(hf + (size_t)arow * 32 + kb);
  const v8h* ap1 = (const v8h*)(hf + (size_t)arow * 32 + kb + 16);
  v8h alo = *ap0;
  v8h ahi = *ap1;
  v16h a;
  #pragma unroll
  for (int j = 0; j < 8; ++j) { a[j] = alo[j]; a[8 + j] = ahi[j]; }

  // ---- B tile: 32x16 f16 (K x outch). lane holds col = wave*16+m, K = kb..kb+7 and kb+16..kb+23
  int col = wave * 16 + m;
  v8h blo = *(const v8h*)(wf + (size_t)col * 32 + kb);
  v8h bhi = *(const v8h*)(wf + (size_t)col * 32 + kb + 16);
  v16h b;
  #pragma unroll
  for (int j = 0; j < 8; ++j) { b[j] = blo[j]; b[8 + j] = bhi[j]; }

  v8f c = {};
  c = __builtin_amdgcn_wmma_f32_16x16x32_f16(
      /*neg_a=*/false, a, /*neg_b=*/false, b,
      /*c_mod=*/(short)0, c, /*reuse_a=*/false, /*reuse_b=*/false);

  // ---- epilogue: bias + relu + atomic mean-pool accumulation.
  // Lane's 8 D rows are contiguous: rbase..rbase+7 -> batch indices via 2x b128 loads.
  float bcol = bias[col];
  int rbase = row0 + 8 * hi;
  if (rbase + 8 <= N) {                 // uniform fast path (always taken: N % 16 == 0)
    int4 b0 = *(const int4*)(batch + rbase);
    int4 b1 = *(const int4*)(batch + rbase + 4);
    int bi[8] = { b0.x, b0.y, b0.z, b0.w, b1.x, b1.y, b1.z, b1.w };
    #pragma unroll
    for (int r = 0; r < 8; ++r) {
      float v = c[r] + bcol;
      v = v > 0.0f ? v : 0.0f;
      unsafeAtomicAdd(&pool[(size_t)bi[r] * HID + col], v);
    }
  } else {
    for (int r = 0; r < 8; ++r) {
      int row = rbase + r;
      if (row < N) {
        float v = c[r] + bcol;
        v = v > 0.0f ? v : 0.0f;
        unsafeAtomicAdd(&pool[(size_t)batch[row] * HID + col], v);
      }
    }
  }
}

// classifier: per-graph block (64 threads): h_avg -> relu(w1) -> w2 -> logits
__global__ void classifier_kernel(const float* __restrict__ pool,
                                  const float* __restrict__ counts,
                                  const float* __restrict__ w1, const float* __restrict__ b1,
                                  const float* __restrict__ w2, const float* __restrict__ b2,
                                  float* __restrict__ out) {
  __shared__ float havg[64];
  __shared__ float h1s[32];
  int g = blockIdx.x;
  int t = threadIdx.x;                  // 64 threads
  float cnt = counts[g];
  cnt = cnt > 1.0f ? cnt : 1.0f;
  havg[t] = pool[(size_t)g * HID + t] / cnt;
  __syncthreads();
  if (t < 32) {
    float acc = b1[t];
    #pragma unroll 8
    for (int k = 0; k < 64; ++k) acc += havg[k] * w1[t * 64 + k];
    h1s[t] = acc > 0.0f ? acc : 0.0f;
  }
  __syncthreads();
  if (t < 16) {
    float acc = b2[t];
    #pragma unroll 8
    for (int k = 0; k < 32; ++k) acc += h1s[k] * w2[t * 32 + k];
    out[(size_t)g * 16 + t] = acc;
  }
}

// ---------------- host launch ----------------
static inline int cdiv(int a, int b) { return (a + b - 1) / b; }

extern "C" void kernel_launch(void* const* d_in, const int* in_sizes, int n_in,
                              void* d_out, int out_size, void* d_ws, size_t ws_size,
                              hipStream_t stream) {
  const float* x       = (const float*)d_in[0];   // [N,32]
  const int*   eidx    = (const int*)  d_in[1];   // [2,E]
  const int*   batch   = (const int*)  d_in[2];   // [N]
  const float* gcn_w   = (const float*)d_in[3];   // [64,32]
  const float* gcn_b   = (const float*)d_in[4];   // [64]
  const float* w1      = (const float*)d_in[5];   // [32,64]
  const float* b1      = (const float*)d_in[6];   // [32]
  const float* w2      = (const float*)d_in[7];   // [16,32]
  const float* b2      = (const float*)d_in[8];   // [16]
  float*       out     = (float*)d_out;

  const int N = in_sizes[0] / 32;
  const int E = in_sizes[1] / 2;
  const int* src = eidx;
  const int* dst = eidx + E;
  const int NC = N * 32;

  // workspace carve-up (256B aligned)
  char* ws = (char*)d_ws;
  size_t off = 0;
  auto alloc = [&](size_t bytes) -> char* {
    char* p = ws + off;
    off += (bytes + 255) & ~(size_t)255;
    return p;
  };
  float*    nrm    = (float*)   alloc((size_t)N * 4);
  float*    hA     = (float*)   alloc((size_t)NC * 4);
  float*    hB     = (float*)   alloc((size_t)NC * 4);
  float*    agg    = (float*)   alloc((size_t)NC * 4);
  _Float16* hf16   = (_Float16*)alloc((size_t)NC * 2);
  _Float16* wf16   = (_Float16*)alloc((size_t)64 * 32 * 2);
  float*    pool   = (float*)   alloc((size_t)N_GRAPHS * HID * 4);
  float*    counts = (float*)   alloc((size_t)N_GRAPHS * 4);
  (void)ws_size;

  const int T = 256;

  // degrees -> norm
  zero_f32<<<cdiv(N, T), T, 0, stream>>>(nrm, N);
  deg_kernel<<<cdiv(E, T), T, 0, stream>>>(dst, nrm, E);
  norm_kernel<<<cdiv(N, T), T, 0, stream>>>(nrm, N);

  // 3 propagation hops (ping-pong: x -> hA -> hB -> hA)
  const float* hop_in[3]  = { x,  hA, hB };
  float*       hop_out[3] = { hA, hB, hA };
  for (int hop = 0; hop < 3; ++hop) {
    zero_f32<<<cdiv(NC, T), T, 0, stream>>>(agg, NC);
    scatter_kernel<<<cdiv(E * 4, T), T, 0, stream>>>(src, dst, nrm, hop_in[hop], agg, E);
    combine_kernel<<<cdiv(NC, T), T, 0, stream>>>(nrm, agg, hop_in[hop], hop_out[hop], NC);
  }

  // convert to f16 for WMMA
  cvt_f16_kernel<<<cdiv(NC, T), T, 0, stream>>>(hA, hf16, NC);
  cvt_f16_kernel<<<cdiv(64 * 32, T), T, 0, stream>>>(gcn_w, wf16, 64 * 32);

  // pooling prep
  zero_f32<<<cdiv(N_GRAPHS * HID, T), T, 0, stream>>>(pool, N_GRAPHS * HID);
  zero_f32<<<1, T, 0, stream>>>(counts, N_GRAPHS);
  counts_kernel<<<cdiv(N, T), T, 0, stream>>>(batch, counts, N);

  // fused WMMA GEMM + bias + relu + mean-pool-sum
  gcn_gemm_relu_pool<<<cdiv(N, 16), 128, 0, stream>>>(hf16, wf16, gcn_b, batch, pool, N);

  // classifier
  classifier_kernel<<<N_GRAPHS, 64, 0, stream>>>(pool, counts, w1, b1, w2, b2, out);
  (void)out_size; (void)n_in;
}
